// GAT_15676630630491
// MI455X (gfx1250) — matrix operand order repaired and checked by reading, not compiled, not run on hardware
//
#include <hip/hip_runtime.h>

// ---------------------------------------------------------------------------
// CDNA5 (gfx1250) GAT forward: WMMA f16 GEMMs (pre-swizzled B fragments) +
// atomic segment softmax/scatter with LDS alpha broadcast.
// ---------------------------------------------------------------------------

typedef _Float16 half16 __attribute__((ext_vector_type(16)));
typedef float    float8 __attribute__((ext_vector_type(8)));

static __device__ __forceinline__ float lrelu02(float x) { return x > 0.f ? x : 0.2f * x; }

// Monotonic float atomic-max via sign-partitioned int/uint atomics.
static __device__ __forceinline__ void atomic_max_f32(float* addr, float v) {
  if (v >= 0.f) atomicMax((int*)addr, __float_as_int(v));
  else          atomicMin((unsigned int*)addr, __float_as_uint(v));
}

// Edge e in [0, E+N): first E from edge_index[2,E], then N implicit self-loops.
static __device__ __forceinline__ void edge_sd(const int* __restrict__ ei, int E,
                                               long long e, int& s, int& d) {
  if (e < (long long)E) { s = ei[e]; d = ei[(long long)E + e]; }
  else                  { int i = (int)(e - (long long)E); s = i; d = i; }
}

// ---------------------------------------------------------------------------
// Pack B[K,N] (f32 row-major) into per-lane WMMA B-fragment layout (f16):
// Bp[((kt*numCT + ct)*32 + lane)*16 + j], lane's 16 halves contiguous, where
//   n = ct*16 + (lane&15),  k = kt*32 + (lane>>4)*16 + 2*(j>>1) + (j&1)
// One thread per (tile, lane): writes 32 contiguous bytes (coalesced).
// ---------------------------------------------------------------------------
__global__ void gatw_pack_b(const float* __restrict__ B, _Float16* __restrict__ Bp,
                            int N, int K) {
  const int numCT = N >> 4;
  const int tid = blockIdx.x * blockDim.x + threadIdx.x;
  const int tot = (K >> 5) * numCT * 32;
  if (tid >= tot) return;
  const int lane = tid & 31;
  const int tile = tid >> 5;           // kt*numCT + ct
  const int ct = tile % numCT;
  const int kt = tile / numCT;
  const int n = (ct << 4) + (lane & 15);
  const int kbase = (kt << 5) + ((lane >> 4) << 4);
  _Float16* dst = Bp + (size_t)tid * 16;
#pragma unroll
  for (int v = 0; v < 8; ++v) {
    const int k = kbase + (v << 1);
    dst[2 * v]     = (_Float16)B[(size_t)k * N + n];
    dst[2 * v + 1] = (_Float16)B[(size_t)(k + 1) * N + n];
  }
}

// ---------------------------------------------------------------------------
// WMMA GEMM: C[M,N] = act(A[M,K] @ B[K,N] + bias), B pre-packed (see above).
// One wave owns ROWS consecutive 16x16 C tiles in M (B fragment reused ROWS x).
// 4 waves/block cover 64 columns. Requires M % (16*ROWS)==0, N%16==0, K%32==0.
// Fragment layouts used here (wave32; A and Bp share a contiguous k order):
//   A 16x32 f16: lane holds row m=lane&15; VGPR v holds K pair
//                k0 = ((lane>>4)<<4) + (v<<1)
//   C 16x16 f32: VGPR r -> row r + (lane>>4)*8, col lane&15
// ---------------------------------------------------------------------------
template <int ROWS, bool BIAS, int ACT>
__global__ void gatw_wmma_gemm(const float* __restrict__ A, const half16* __restrict__ Bp,
                               const float* __restrict__ bias, float* __restrict__ C,
                               int M, int N, int K) {
  const int lane    = threadIdx.x & 31;
  const int wave    = threadIdx.x >> 5;
  const int colTile = blockIdx.y * 4 + wave;
  const int numCT   = N >> 4;
  const int n0      = colTile * 16;
  const int m0      = blockIdx.x * (16 * ROWS);
  if (n0 >= N) return;
  const int lh   = lane & 15;
  const int hfm  = lane >> 4;              // lane half (0/1)
  const int bCol = n0 + lh;

  const float8 zero = {0.f, 0.f, 0.f, 0.f, 0.f, 0.f, 0.f, 0.f};
  float8 acc[ROWS];
#pragma unroll
  for (int rr = 0; rr < ROWS; ++rr) acc[rr] = zero;

  const float* Abase = A + (size_t)(m0 + lh) * K;

  for (int kk = 0, kt = 0; kk < K; kk += 32, ++kt) {
    const half16 bfrag = Bp[((size_t)kt * numCT + colTile) * 32 + lane];
#pragma unroll
    for (int rr = 0; rr < ROWS; ++rr) {
      const float* Ar = Abase + (size_t)rr * 16 * K;
      half16 afrag;
#pragma unroll
      for (int v = 0; v < 8; ++v) {
        const int ka = kk + (hfm << 4) + (v << 1);
        const float2 av = *(const float2*)(Ar + ka);
        afrag[2 * v]     = (_Float16)av.x;
        afrag[2 * v + 1] = (_Float16)av.y;
      }
      acc[rr] = __builtin_amdgcn_wmma_f32_16x16x32_f16(false, afrag, false, bfrag,
                                                       (short)0, acc[rr], false, false);
    }
  }

  const float bv = BIAS ? bias[bCol] : 0.f;
#pragma unroll
  for (int rr = 0; rr < ROWS; ++rr) {
    float* Cp = C + (size_t)(m0 + rr * 16 + hfm * 8) * N + bCol;
#pragma unroll
    for (int r = 0; r < 8; ++r) {
      float v = acc[rr][r] + bv;
      if (ACT == 1) v = fmaxf(v, 0.f);   // ReLU
      Cp[(size_t)r * N] = v;
    }
  }
}

// ---------------------------------------------------------------------------
// Attention logits: als[n,h] = sum_c h[n,h*C+c]*a_s[h,c]; same for ald.
// ---------------------------------------------------------------------------
__global__ void gatw_attn_logits(const float* __restrict__ h,
                                 const float* __restrict__ a_s, const float* __restrict__ a_d,
                                 float* __restrict__ als, float* __restrict__ ald,
                                 int n, int Hh, int C) {
  const long long tid = (long long)blockIdx.x * blockDim.x + threadIdx.x;
  if (tid >= (long long)n * Hh) return;
  const int hh   = (int)(tid % Hh);
  const long long node = tid / Hh;
  const float* hp = h + node * Hh * C + (long long)hh * C;
  const float* as = a_s + hh * C;
  const float* ad = a_d + hh * C;
  float ss = 0.f, sd = 0.f;
  for (int c = 0; c < C; ++c) { const float x = hp[c]; ss += x * as[c]; sd += x * ad[c]; }
  als[tid] = ss;
  ald[tid] = sd;
}

// Init per-layer scratch: mx = -inf, den = 0 over n*Hh; acc = 0 over n*Hh*C.
__global__ void gatw_attn_init(float* __restrict__ mx, float* __restrict__ den,
                               float* __restrict__ acc, long long nH, long long nHC) {
  const long long tid = (long long)blockIdx.x * blockDim.x + threadIdx.x;
  if (tid < nH)  { mx[tid] = -__builtin_inff(); den[tid] = 0.f; }
  if (tid < nHC) acc[tid] = 0.f;
}

__global__ void gatw_attn_max(const int* __restrict__ ei, int E, int Nn,
                              const float* __restrict__ als, const float* __restrict__ ald,
                              float* __restrict__ mx, int Hh) {
  const long long tid = (long long)blockIdx.x * blockDim.x + threadIdx.x;
  const long long tot = (long long)(E + Nn) * Hh;
  if (tid >= tot) return;
  const int hh = (int)(tid % Hh);
  int s, d; edge_sd(ei, E, tid / Hh, s, d);
  const float ev = lrelu02(als[(long long)s * Hh + hh] + ald[(long long)d * Hh + hh]);
  atomic_max_f32(&mx[(long long)d * Hh + hh], ev);
}

__global__ void gatw_attn_den(const int* __restrict__ ei, int E, int Nn,
                              const float* __restrict__ als, const float* __restrict__ ald,
                              const float* __restrict__ mx, float* __restrict__ den, int Hh) {
  const long long tid = (long long)blockIdx.x * blockDim.x + threadIdx.x;
  const long long tot = (long long)(E + Nn) * Hh;
  if (tid >= tot) return;
  const int hh = (int)(tid % Hh);
  int s, d; edge_sd(ei, E, tid / Hh, s, d);
  const long long di = (long long)d * Hh + hh;
  const float ev = lrelu02(als[(long long)s * Hh + hh] + ald[di]);
  atomicAdd(&den[di], expf(ev - mx[di]));
}

// One block per edge; thread t covers feature t = hh*C + c of Hh*C features.
// alpha computed once per (edge, head) in the first Hh threads, broadcast
// through LDS; remaining threads only gather-multiply-atomicAdd.
__global__ void gatw_attn_agg(const int* __restrict__ ei, int E, int Nn,
                              const float* __restrict__ als, const float* __restrict__ ald,
                              const float* __restrict__ mx, const float* __restrict__ den,
                              const float* __restrict__ hfeat, float* __restrict__ acc,
                              int Hh, int C) {
  __shared__ float sAlpha[8];
  const long long e = blockIdx.x;
  const int t = threadIdx.x;
  int s, d; edge_sd(ei, E, e, s, d);
  if (t < Hh) {
    const long long di = (long long)d * Hh + t;
    const float ev = lrelu02(als[(long long)s * Hh + t] + ald[di]);
    sAlpha[t] = expf(ev - mx[di]) / den[di];
  }
  __syncthreads();
  const float alpha = sAlpha[t / C];
  const long long F = (long long)Hh * C;
  atomicAdd(&acc[(long long)d * F + t], alpha * hfeat[(long long)s * F + t]);
}

// y = elu((x + b - bm) * rsqrt(bv+eps) * g + bb), in place.
__global__ void gatw_bn_elu(float* __restrict__ acc, const float* __restrict__ b,
                            const float* __restrict__ g, const float* __restrict__ bb,
                            const float* __restrict__ bm, const float* __restrict__ bv,
                            long long n, int F) {
  const long long tid = (long long)blockIdx.x * blockDim.x + threadIdx.x;
  if (tid >= n * F) return;
  const int c = (int)(tid % F);
  float x = acc[tid] + b[c];
  x = (x - bm[c]) * rsqrtf(bv[c] + 1e-5f) * g[c] + bb[c];
  acc[tid] = x > 0.f ? x : (expf(x) - 1.f);
}

__global__ void gatw_add_bias(float* __restrict__ acc, const float* __restrict__ b,
                              long long n, int F) {
  const long long tid = (long long)blockIdx.x * blockDim.x + threadIdx.x;
  if (tid >= n * F) return;
  acc[tid] += b[(int)(tid % F)];
}

// ---------------------------------------------------------------------------
// Graph pooling (G=64, C=64)
// ---------------------------------------------------------------------------
__global__ void gatw_pool_init(float* __restrict__ pmax, float* __restrict__ psum,
                               float* __restrict__ cnt, int GC, int G) {
  const int tid = blockIdx.x * blockDim.x + threadIdx.x;
  if (tid < GC) { pmax[tid] = -__builtin_inff(); psum[tid] = 0.f; }
  if (tid < G)  cnt[tid] = 0.f;
}

__global__ void gatw_pool_cnt(const int* __restrict__ batch, float* __restrict__ cnt, int n) {
  const int tid = blockIdx.x * blockDim.x + threadIdx.x;
  if (tid >= n) return;
  atomicAdd(&cnt[batch[tid]], 1.f);
}

__global__ void gatw_pool_scatter(const float* __restrict__ h3, const int* __restrict__ batch,
                                  float* __restrict__ pmax, float* __restrict__ psum,
                                  int n, int C) {
  const long long tid = (long long)blockIdx.x * blockDim.x + threadIdx.x;
  if (tid >= (long long)n * C) return;
  const int c = (int)(tid % C);
  const int node = (int)(tid / C);
  const int g = batch[node];
  const float v = h3[tid];
  atomic_max_f32(&pmax[g * C + c], v);
  atomicAdd(&psum[g * C + c], v);
}

__global__ void gatw_build_z(const float* __restrict__ pmax, const float* __restrict__ psum,
                             const float* __restrict__ cnt, float* __restrict__ z,
                             int G, int C) {
  const int tid = blockIdx.x * blockDim.x + threadIdx.x;
  if (tid >= G * 2 * C) return;
  const int g = tid / (2 * C);
  const int c = tid % (2 * C);
  z[tid] = (c < C) ? pmax[g * C + c]
                   : psum[g * C + (c - C)] / fmaxf(cnt[g], 1.f);
}

// ---------------------------------------------------------------------------
static inline int cdiv(long long a, int b) { return (int)((a + b - 1) / b); }

extern "C" void kernel_launch(void* const* d_in, const int* in_sizes, int n_in,
                              void* d_out, int out_size, void* d_ws, size_t ws_size,
                              hipStream_t stream) {
  (void)n_in; (void)out_size; (void)ws_size;

  const float* x     = (const float*)d_in[0];
  const int*   ei    = (const int*)  d_in[1];
  const int*   batch = (const int*)  d_in[2];
  const float* W1  = (const float*)d_in[3];
  const float* as1 = (const float*)d_in[4];
  const float* ad1 = (const float*)d_in[5];
  const float* b1  = (const float*)d_in[6];
  const float* bn1g = (const float*)d_in[7];
  const float* bn1b = (const float*)d_in[8];
  const float* bn1m = (const float*)d_in[9];
  const float* bn1v = (const float*)d_in[10];
  const float* W2  = (const float*)d_in[11];
  const float* as2 = (const float*)d_in[12];
  const float* ad2 = (const float*)d_in[13];
  const float* b2  = (const float*)d_in[14];
  const float* bn2g = (const float*)d_in[15];
  const float* bn2b = (const float*)d_in[16];
  const float* bn2m = (const float*)d_in[17];
  const float* bn2v = (const float*)d_in[18];
  const float* W3  = (const float*)d_in[19];
  const float* as3 = (const float*)d_in[20];
  const float* ad3 = (const float*)d_in[21];
  const float* b3  = (const float*)d_in[22];
  const float* P1  = (const float*)d_in[23];
  const float* pb1 = (const float*)d_in[24];
  const float* P2  = (const float*)d_in[25];
  const float* pb2 = (const float*)d_in[26];

  const int Nn  = in_sizes[2];                 // 50000 nodes
  const int E   = in_sizes[1] / 2;             // 800000 edges
  const int IN  = (int)((long long)in_sizes[0] / Nn);   // 128
  const int HID = 64, H = 4, G = 64;
  const int HC  = H * HID;                     // 256
  const int Ep  = E + Nn;                      // with self loops

  // Workspace layout (floats); total ~106 MB, L2-resident on MI455X (192 MB).
  float* ws = (float*)d_ws;
  size_t off = 0;
  float* hfeat = ws + off; off += (size_t)Nn * HC;   // pre-aggregation features
  float* acc   = ws + off; off += (size_t)Nn * HC;   // aggregation target / layer io
  float* als   = ws + off; off += (size_t)Nn * H;
  float* ald   = ws + off; off += (size_t)Nn * H;
  float* mx    = ws + off; off += (size_t)Nn * H;
  float* den   = ws + off; off += (size_t)Nn * H;
  float* pmax  = ws + off; off += (size_t)G * HID;
  float* psum  = ws + off; off += (size_t)G * HID;
  float* cnt   = ws + off; off += (size_t)G;
  float* z     = ws + off; off += (size_t)G * 2 * HID;
  float* t1    = ws + off; off += (size_t)G * HID;
  // f16 packed-weight region (32-byte aligned: off is a multiple of 8 floats)
  off = (off + 7) & ~(size_t)7;
  _Float16* wp = (_Float16*)(ws + off);
  size_t hoff = 0;
  _Float16* W1p = wp + hoff; hoff += (size_t)IN * HC;        // 128x256
  _Float16* W2p = wp + hoff; hoff += (size_t)HC * HC;        // 256x256
  _Float16* W3p = wp + hoff; hoff += (size_t)HC * HID;       // 256x64
  _Float16* P1p = wp + hoff; hoff += (size_t)2 * HID * HID;  // 128x64
  _Float16* P2p = wp + hoff; hoff += (size_t)HID * HID;      // 64x64

  const int TB = 256;
  const int ROWS = 5;                    // 50000 = (16*5)*625
  const dim3 gemmBlk(128);

  // ---------------- Pack all weight matrices into WMMA B-fragment layout ----
  gatw_pack_b<<<cdiv((IN / 32) * (HC / 16) * 32, TB), TB, 0, stream>>>(W1, W1p, HC, IN);
  gatw_pack_b<<<cdiv((HC / 32) * (HC / 16) * 32, TB), TB, 0, stream>>>(W2, W2p, HC, HC);
  gatw_pack_b<<<cdiv((HC / 32) * (HID / 16) * 32, TB), TB, 0, stream>>>(W3, W3p, HID, HC);
  gatw_pack_b<<<cdiv((2 * HID / 32) * (HID / 16) * 32, TB), TB, 0, stream>>>(P1, P1p, HID, 2 * HID);
  gatw_pack_b<<<cdiv((HID / 32) * (HID / 16) * 32, TB), TB, 0, stream>>>(P2, P2p, HID, HID);

  // ---------------- Layer 1: x[N,128] -> h1[N,256], H=4 ----------------
  gatw_wmma_gemm<5, false, 0><<<dim3(Nn / (16 * ROWS), HC / 64), gemmBlk, 0, stream>>>(
      x, (const half16*)W1p, nullptr, hfeat, Nn, HC, IN);
  gatw_attn_logits<<<cdiv((long long)Nn * H, TB), TB, 0, stream>>>(hfeat, as1, ad1, als, ald, Nn, H, HID);
  gatw_attn_init<<<cdiv((long long)Nn * HC, TB), TB, 0, stream>>>(mx, den, acc, (long long)Nn * H, (long long)Nn * HC);
  gatw_attn_max<<<cdiv((long long)Ep * H, TB), TB, 0, stream>>>(ei, E, Nn, als, ald, mx, H);
  gatw_attn_den<<<cdiv((long long)Ep * H, TB), TB, 0, stream>>>(ei, E, Nn, als, ald, mx, den, H);
  gatw_attn_agg<<<Ep, H * HID, 0, stream>>>(ei, E, Nn, als, ald, mx, den, hfeat, acc, H, HID);
  gatw_bn_elu<<<cdiv((long long)Nn * HC, TB), TB, 0, stream>>>(acc, b1, bn1g, bn1b, bn1m, bn1v, Nn, HC);

  // ---------------- Layer 2: h1[N,256] -> h2[N,256], H=4 ----------------
  gatw_wmma_gemm<5, false, 0><<<dim3(Nn / (16 * ROWS), HC / 64), gemmBlk, 0, stream>>>(
      acc, (const half16*)W2p, nullptr, hfeat, Nn, HC, HC);
  gatw_attn_logits<<<cdiv((long long)Nn * H, TB), TB, 0, stream>>>(hfeat, as2, ad2, als, ald, Nn, H, HID);
  gatw_attn_init<<<cdiv((long long)Nn * HC, TB), TB, 0, stream>>>(mx, den, acc, (long long)Nn * H, (long long)Nn * HC);
  gatw_attn_max<<<cdiv((long long)Ep * H, TB), TB, 0, stream>>>(ei, E, Nn, als, ald, mx, H);
  gatw_attn_den<<<cdiv((long long)Ep * H, TB), TB, 0, stream>>>(ei, E, Nn, als, ald, mx, den, H);
  gatw_attn_agg<<<Ep, H * HID, 0, stream>>>(ei, E, Nn, als, ald, mx, den, hfeat, acc, H, HID);
  gatw_bn_elu<<<cdiv((long long)Nn * HC, TB), TB, 0, stream>>>(acc, b2, bn2g, bn2b, bn2m, bn2v, Nn, HC);

  // ---------------- Layer 3: h2[N,256] -> h3[N,64], H=1 ----------------
  gatw_wmma_gemm<5, false, 0><<<dim3(Nn / (16 * ROWS), 1), gemmBlk, 0, stream>>>(
      acc, (const half16*)W3p, nullptr, hfeat, Nn, HID, HC);
  gatw_attn_logits<<<cdiv((long long)Nn, TB), TB, 0, stream>>>(hfeat, as3, ad3, als, ald, Nn, 1, HID);
  gatw_attn_init<<<cdiv((long long)Nn * HID, TB), TB, 0, stream>>>(mx, den, acc, (long long)Nn, (long long)Nn * HID);
  gatw_attn_max<<<cdiv((long long)Ep, TB), TB, 0, stream>>>(ei, E, Nn, als, ald, mx, 1);
  gatw_attn_den<<<cdiv((long long)Ep, TB), TB, 0, stream>>>(ei, E, Nn, als, ald, mx, den, 1);
  gatw_attn_agg<<<Ep, HID, 0, stream>>>(ei, E, Nn, als, ald, mx, den, hfeat, acc, 1, HID);
  gatw_add_bias<<<cdiv((long long)Nn * HID, TB), TB, 0, stream>>>(acc, b3, Nn, HID);

  // ---------------- Pooling + projection MLP ----------------
  gatw_pool_init<<<cdiv(G * HID, TB), TB, 0, stream>>>(pmax, psum, cnt, G * HID, G);
  gatw_pool_cnt<<<cdiv(Nn, TB), TB, 0, stream>>>(batch, cnt, Nn);
  gatw_pool_scatter<<<cdiv((long long)Nn * HID, TB), TB, 0, stream>>>(acc, batch, pmax, psum, Nn, HID);
  gatw_build_z<<<cdiv(G * 2 * HID, TB), TB, 0, stream>>>(pmax, psum, cnt, z, G, HID);

  // z[64,128] @ P1[128,64] + pb1, ReLU -> t1 ; t1 @ P2[64,64] + pb2 -> out
  gatw_wmma_gemm<4, true, 1><<<dim3(1, 1), gemmBlk, 0, stream>>>(
      z, (const half16*)P1p, pb1, t1, G, HID, 2 * HID);
  gatw_wmma_gemm<4, true, 0><<<dim3(1, 1), gemmBlk, 0, stream>>>(
      t1, (const half16*)P2p, pb2, (float*)d_out, G, HID, HID);
}
